// DecoderRNN_71966472012390
// MI455X (gfx1250) — matrix operand order repaired and checked by reading, not compiled, not run on hardware
//
#include <hip/hip_runtime.h>

// ---------------------------------------------------------------------------
// DecoderRNN (LSTM caption decoder) for MI455X / gfx1250, wave32 + WMMA bf16.
//   B=64, T=32, E=H=512, V=16000, G=4H=2048.
// Pipeline:
//   prep        : fp32->bf16 weight conversion, x_seq gather, bias sum, zeros
//   gemm_xproj  : x_proj[t*B+b][g] = x_seq @ W_ih^T + (b_ih+b_hh)  (WMMA bf16)
//   lstm_step x32: gates = x_proj[t] + h @ W_hh^T -> fused i/f/g/o epilogue
//   gemm_logits : out = hs @ W_lin^T + b_lin                       (WMMA bf16)
// Matmuls in bf16 with f32 accumulation; c-state kept in f32.
// __launch_bounds__(256,1) releases the VGPR cap so the 4x4 accumulator
// macro-tile (128 VGPRs) lives entirely in registers (no scratch spills).
// ---------------------------------------------------------------------------

typedef __attribute__((ext_vector_type(16))) __bf16 v16bf;
typedef __attribute__((ext_vector_type(8)))  __bf16 v8bf;
typedef __attribute__((ext_vector_type(8)))  float  v8f;

#define WMMA_BF16(a, b, c) \
  __builtin_amdgcn_wmma_f32_16x16x32_bf16(false, (a), false, (b), (short)0, (c), false, false)

static constexpr int Bc = 64;
static constexpr int Tc = 32;
static constexpr int Ec = 512;
static constexpr int Hc = 512;
static constexpr int Vc = 16000;
static constexpr int Gc = 4 * Hc;   // 2048 gate columns
static constexpr int BT = Bc * Tc;  // 2048 rows (t-major: row = t*B + b)

// --- WMMA fragment loaders (contiguous k order, same for A and B, wave32) -------

// A: 16x32 bf16 (MxK). lane&15 = row M; lanes>=16 take K offset +16.
// element e -> K = e + 16*(lane>>4). Two contiguous 16B loads.
__device__ __forceinline__ v16bf load_a_frag(const __bf16* A, int lda,
                                             int Mbase, int kb, int lane) {
  int row  = Mbase + (lane & 15);
  int half = lane >> 4;
  const __bf16* p = A + (size_t)row * lda + kb + 16 * half;
  v8bf lo = *(const v8bf*)(p);
  v8bf hi = *(const v8bf*)(p + 8);
  return __builtin_shufflevector(lo, hi, 0, 1, 2, 3, 4, 5, 6, 7,
                                 8, 9, 10, 11, 12, 13, 14, 15);
}

// B: 32x16 bf16 (KxN). Our B-operand is W^T with W stored [N][K] row-major:
// lane&15 = column N (= W row); element e -> K = e + 16*(lane>>4).
// One contiguous 32B load from W's row.
__device__ __forceinline__ v16bf load_b_frag(const __bf16* W, int ldw,
                                             int Nbase, int kb, int lane) {
  int col  = Nbase + (lane & 15);
  int half = lane >> 4;
  const __bf16* p = W + (size_t)col * ldw + kb + 16 * half;
  return *(const v16bf*)p;
}

__device__ __forceinline__ float sigmoidf_fast(float x) {
  return 1.0f / (1.0f + __expf(-x));
}

// --- prep kernels -----------------------------------------------------------

__global__ void convert_bf16_kernel(const float* __restrict__ src,
                                    __bf16* __restrict__ dst, int n) {
  for (int i = blockIdx.x * blockDim.x + threadIdx.x; i < n;
       i += gridDim.x * blockDim.x)
    dst[i] = (__bf16)src[i];
}

// x_seq row r = t*B + b: t==0 -> features[b], t>=1 -> emb[captions[b][t]]
__global__ void build_xseq_kernel(const float* __restrict__ features,
                                  const int* __restrict__ captions,
                                  const float* __restrict__ emb,
                                  __bf16* __restrict__ xseq) {
  const int n = BT * Ec;
  for (int i = blockIdx.x * blockDim.x + threadIdx.x; i < n;
       i += gridDim.x * blockDim.x) {
    int r = i >> 9;         // row, Ec==512
    int k = i & 511;
    int t = r >> 6;         // Bc==64
    int b = r & 63;
    float v;
    if (t == 0) {
      v = features[b * Ec + k];
    } else {
      int cap = captions[b * Tc + t];
      v = emb[(size_t)cap * Ec + k];
    }
    xseq[i] = (__bf16)v;
  }
}

// bsum = b_ih + b_hh; zero c-state and both h double-buffers (h0 = 0).
__global__ void init_misc_kernel(const float* __restrict__ b_ih,
                                 const float* __restrict__ b_hh,
                                 float* __restrict__ bsum,
                                 float* __restrict__ cstate,
                                 __bf16* __restrict__ hbuf) {
  int i = blockIdx.x * blockDim.x + threadIdx.x;  // 65536 threads
  if (i < Gc) bsum[i] = b_ih[i] + b_hh[i];
  if (i < Bc * Hc) cstate[i] = 0.0f;
  hbuf[i] = (__bf16)0.0f;                         // 2 * Bc * Hc entries
}

// --- GEMM 1: x_proj = x_seq @ W_ih^T + bsum  (M=2048, N=2048, K=512) --------
// One wave computes a 64x64 macro-tile (4x4 WMMA tiles). 1024 waves total.

__global__ __launch_bounds__(256, 1)
void gemm_xproj_kernel(const __bf16* __restrict__ A,   // [BT][Ec]
                       const __bf16* __restrict__ W,   // [Gc][Ec]
                       const float* __restrict__ bsum, // [Gc]
                       float* __restrict__ C) {        // [BT][Gc]
  const int lane   = threadIdx.x & 31;
  const int waveId = (blockIdx.x * blockDim.x + threadIdx.x) >> 5;
  const int nWaveN = Gc / 64;             // 32
  const int Mbase  = (waveId / nWaveN) * 64;
  const int Nbase  = (waveId % nWaveN) * 64;

  v8f acc[4][4] = {};
  for (int kb = 0; kb < Ec; kb += 32) {
    v16bf a[4], b[4];
#pragma unroll
    for (int i = 0; i < 4; ++i) a[i] = load_a_frag(A, Ec, Mbase + 16 * i, kb, lane);
#pragma unroll
    for (int j = 0; j < 4; ++j) b[j] = load_b_frag(W, Ec, Nbase + 16 * j, kb, lane);
#pragma unroll
    for (int i = 0; i < 4; ++i)
#pragma unroll
      for (int j = 0; j < 4; ++j) acc[i][j] = WMMA_BF16(a[i], b[j], acc[i][j]);
  }
  const int rofs = 8 * (lane >> 4);
  const int cofs = lane & 15;
#pragma unroll
  for (int i = 0; i < 4; ++i)
#pragma unroll
    for (int j = 0; j < 4; ++j) {
      int col = Nbase + 16 * j + cofs;
      float bias = bsum[col];
      float* cp = C + (size_t)(Mbase + 16 * i + rofs) * Gc + col;
#pragma unroll
      for (int r = 0; r < 8; ++r) cp[(size_t)r * Gc] = acc[i][j][r] + bias;
    }
}

// --- LSTM step: gates = x_proj[t] + h_in @ W_hh^T, fused ifgo epilogue ------
// M=64, K=512. Each wave owns one (16-row, 16-hcol) patch and computes all
// four gate tiles for it (N = g*512 + jt*16), so the nonlinearities and the
// c/h update stay in registers. 128 waves = 16 blocks x 8 waves.

__global__ __launch_bounds__(256, 1)
void lstm_step_kernel(const __bf16* __restrict__ h_in,  // [Bc][Hc]
                      __bf16* __restrict__ h_out,       // [Bc][Hc]
                      float* __restrict__ cstate,       // [Bc][Hc]
                      const __bf16* __restrict__ Whh,   // [Gc][Hc]
                      const float* __restrict__ xproj,  // [BT][Gc]
                      __bf16* __restrict__ hs,          // [BT][Hc]
                      int t) {
  const int lane   = threadIdx.x & 31;
  const int waveId = (blockIdx.x * blockDim.x + threadIdx.x) >> 5;
  const int mt = waveId >> 5;        // 0..3   (batch tile)
  const int jt = waveId & 31;        // 0..31  (h-column tile)
  const int Mbase = mt * 16;
  const int rofs  = 8 * (lane >> 4);
  const int cofs  = lane & 15;

  // Init accumulators from precomputed input projection (includes biases).
  v8f acc[4];
#pragma unroll
  for (int g = 0; g < 4; ++g) {
    const float* xp = xproj + (size_t)(t * Bc + Mbase + rofs) * Gc
                            + g * Hc + jt * 16 + cofs;
#pragma unroll
    for (int r = 0; r < 8; ++r) acc[g][r] = xp[(size_t)r * Gc];
  }

  for (int kb = 0; kb < Hc; kb += 32) {
    v16bf a = load_a_frag(h_in, Hc, Mbase, kb, lane);
#pragma unroll
    for (int g = 0; g < 4; ++g) {
      v16bf b = load_b_frag(Whh, Hc, g * Hc + jt * 16, kb, lane);
      acc[g] = WMMA_BF16(a, b, acc[g]);
    }
  }

  // Fused LSTM pointwise update (PyTorch ifgo gate order).
#pragma unroll
  for (int r = 0; r < 8; ++r) {
    int m   = Mbase + r + rofs;      // batch row 0..63
    int col = jt * 16 + cofs;        // h column 0..511
    float iv = sigmoidf_fast(acc[0][r]);
    float fv = sigmoidf_fast(acc[1][r]);
    float gv = tanhf(acc[2][r]);
    float ov = sigmoidf_fast(acc[3][r]);
    size_t idx = (size_t)m * Hc + col;
    float cn = fv * cstate[idx] + iv * gv;
    cstate[idx] = cn;
    float hv = ov * tanhf(cn);
    __bf16 hb = (__bf16)hv;
    h_out[idx] = hb;
    hs[(size_t)(t * Bc + m) * Hc + col] = hb;
  }
}

// --- GEMM 2: logits = hs @ W_lin^T + b_lin  (M=2048, N=16000, K=512) --------
// Dominant GEMM (33.6 GFLOP). 64x64 macro-tile per wave; 8000 waves.
// Output rows r = t*B + b are remapped to the [B][T][V] layout on store.

__global__ __launch_bounds__(256, 1)
void gemm_logits_kernel(const __bf16* __restrict__ hs,   // [BT][Hc]
                        const __bf16* __restrict__ W,    // [Vc][Hc]
                        const float* __restrict__ b_lin, // [Vc]
                        float* __restrict__ out) {       // [B][T][V]
  const int lane   = threadIdx.x & 31;
  const int waveId = (blockIdx.x * blockDim.x + threadIdx.x) >> 5;
  const int nWaveN = Vc / 64;              // 250
  const int Mbase  = (waveId / nWaveN) * 64;
  const int Nbase  = (waveId % nWaveN) * 64;

  v8f acc[4][4] = {};
  for (int kb = 0; kb < Hc; kb += 32) {
    // Prefetch next K-panel of W_lin (gfx1250 global_prefetch_b8 path).
    if (kb + 32 < Hc)
      __builtin_prefetch(W + (size_t)(Nbase + (lane & 15)) * Hc + kb + 32, 0, 1);
    v16bf a[4], b[4];
#pragma unroll
    for (int i = 0; i < 4; ++i) a[i] = load_a_frag(hs, Hc, Mbase + 16 * i, kb, lane);
#pragma unroll
    for (int j = 0; j < 4; ++j) b[j] = load_b_frag(W, Hc, Nbase + 16 * j, kb, lane);
#pragma unroll
    for (int i = 0; i < 4; ++i)
#pragma unroll
      for (int j = 0; j < 4; ++j) acc[i][j] = WMMA_BF16(a[i], b[j], acc[i][j]);
  }
  const int rofs = 8 * (lane >> 4);
  const int cofs = lane & 15;
#pragma unroll
  for (int i = 0; i < 4; ++i) {
    int row0 = Mbase + 16 * i + rofs;  // = t*B + b
#pragma unroll
    for (int j = 0; j < 4; ++j) {
      int col = Nbase + 16 * j + cofs;
      float bias = b_lin[col];
#pragma unroll
      for (int r = 0; r < 8; ++r) {
        int row = row0 + r;
        int t = row >> 6;
        int b = row & 63;
        out[((size_t)(b * Tc + t)) * Vc + col] = acc[i][j][r] + bias;
      }
    }
  }
}

// ---------------------------------------------------------------------------

extern "C" void kernel_launch(void* const* d_in, const int* in_sizes, int n_in,
                              void* d_out, int out_size, void* d_ws, size_t ws_size,
                              hipStream_t stream) {
  const float* features = (const float*)d_in[0];
  const int*   captions = (const int*)d_in[1];
  const float* emb      = (const float*)d_in[2];
  const float* W_ih     = (const float*)d_in[3];
  const float* W_hh     = (const float*)d_in[4];
  const float* b_ih     = (const float*)d_in[5];
  const float* b_hh     = (const float*)d_in[6];
  const float* W_lin    = (const float*)d_in[7];
  const float* b_lin    = (const float*)d_in[8];
  float* out = (float*)d_out;

  // Workspace carve-up (~42 MB), 256B aligned.
  char* base = (char*)d_ws;
  size_t off = 0;
  auto take = [&](size_t bytes) -> char* {
    char* p = base + off;
    off = (off + bytes + 255) & ~(size_t)255;
    return p;
  };
  __bf16* Wih_bf  = (__bf16*)take((size_t)Gc * Ec * 2);
  __bf16* Whh_bf  = (__bf16*)take((size_t)Gc * Hc * 2);
  __bf16* Wlin_bf = (__bf16*)take((size_t)Vc * Hc * 2);
  __bf16* xseq_bf = (__bf16*)take((size_t)BT * Ec * 2);
  __bf16* hs_bf   = (__bf16*)take((size_t)BT * Hc * 2);
  float*  xproj   = (float*) take((size_t)BT * Gc * 4);
  float*  bsum    = (float*) take((size_t)Gc * 4);
  float*  cstate  = (float*) take((size_t)Bc * Hc * 4);
  __bf16* hbuf    = (__bf16*)take((size_t)2 * Bc * Hc * 2);

  // 1) Weight conversion + input staging.
  convert_bf16_kernel<<<512,  256, 0, stream>>>(W_ih,  Wih_bf,  Gc * Ec);
  convert_bf16_kernel<<<512,  256, 0, stream>>>(W_hh,  Whh_bf,  Gc * Hc);
  convert_bf16_kernel<<<2048, 256, 0, stream>>>(W_lin, Wlin_bf, Vc * Hc);
  build_xseq_kernel<<<1024, 256, 0, stream>>>(features, captions, emb, xseq_bf);
  init_misc_kernel<<<256, 256, 0, stream>>>(b_ih, b_hh, bsum, cstate, hbuf);

  // 2) Input projection for all timesteps (parallel over t).
  gemm_xproj_kernel<<<128, 256, 0, stream>>>(xseq_bf, Wih_bf, bsum, xproj);

  // 3) Sequential LSTM recurrence; stream ordering provides the t->t+1 dep.
  //    h double-buffered so all waves of step t read a stable h_in.
  for (int t = 0; t < Tc; ++t) {
    const __bf16* h_in  = hbuf + (size_t)(t & 1) * Bc * Hc;
    __bf16*       h_out = hbuf + (size_t)((t + 1) & 1) * Bc * Hc;
    lstm_step_kernel<<<16, 256, 0, stream>>>(h_in, h_out, cstate, Whh_bf,
                                             xproj, hs_bf, t);
  }

  // 4) Vocabulary projection (dominant GEMM) with [B][T][V] store remap.
  gemm_logits_kernel<<<1000, 256, 0, stream>>>(hs_bf, Wlin_bf, b_lin, out);
}
